// Decoder_17403207483392
// MI455X (gfx1250) — matrix operand order repaired and checked by reading, not compiled, not run on hardware
//
#include <hip/hip_runtime.h>
#include <hip/hip_bf16.h>

typedef __bf16 bf16_t;
typedef __attribute__((ext_vector_type(16))) __bf16 v16bf;
typedef __attribute__((ext_vector_type(8)))  __bf16 v8bf;
typedef __attribute__((ext_vector_type(8)))  float  v8f;
typedef __attribute__((ext_vector_type(4)))  float  fvec4;   // plain ext-vector (not HIP class)

// Explicit global-address-space pointer types: force global_load/global_store
// (FLAT ops would also tick DScnt and occupy the LDS pipe on CDNA5).
typedef __attribute__((address_space(1))) const v8bf*   gc_v8bf;
typedef __attribute__((address_space(1))) const v16bf*  gc_v16bf;
typedef __attribute__((address_space(1))) const float*  gc_f32;
typedef __attribute__((address_space(1))) const fvec4*  gc_f4;
typedef __attribute__((address_space(1))) float*        g_f32;
typedef __attribute__((address_space(1))) bf16_t*       g_bf16;

#define HID   512
#define BATCH 64
#define SEQ   128
#define TSTEP 32          // T-1
#define VOCABN 32000

// Up to 3 K-sections: each pairs an A slab (rows x len) with a column range of W.
struct GemmSect {
  const bf16_t* A; int lda; int kA;   // A base, row stride, col offset (elements)
  const bf16_t* W; int ldw; int kW;   // W base ([N][K] row-major), row stride, col offset
  int len;                            // K length for this section (multiple of 64)
};

__device__ __forceinline__ void
gemm_section(const GemmSect& s, int row, int l16, int half, int ncol0,
             v8f& acc0, v8f& acc1, v8f& acc2, v8f& acc3)
{
  const bf16_t* Abase = s.A + (size_t)row * s.lda + s.kA;
  const bf16_t* Wbase = s.W + (size_t)(ncol0 + l16) * s.ldw + s.kW + half * 16;
  const int ldw = s.ldw;
#pragma unroll 2
  for (int k = 0; k < s.len; k += 32) {
    // A frag: row = row, lane<16 -> K k..k+15, lane>=16 -> K k+16..k+31
    const bf16_t* ap = Abase + k + half * 16;
    v16bf a = *(gc_v16bf)(ap);
    // B frag: col = ncol+l16, lane<16 -> K k..k+15, lane>=16 -> K k+16..k+31
    const bf16_t* wp = Wbase + k;
    v16bf b0 = *(gc_v16bf)(wp);
    v16bf b1 = *(gc_v16bf)(wp + (size_t)16 * ldw);
    v16bf b2 = *(gc_v16bf)(wp + (size_t)32 * ldw);
    v16bf b3 = *(gc_v16bf)(wp + (size_t)48 * ldw);
    acc0 = __builtin_amdgcn_wmma_f32_16x16x32_bf16(false, a, false, b0, (short)0, acc0, false, false);
    acc1 = __builtin_amdgcn_wmma_f32_16x16x32_bf16(false, a, false, b1, (short)0, acc1, false, false);
    acc2 = __builtin_amdgcn_wmma_f32_16x16x32_bf16(false, a, false, b2, (short)0, acc2, false, false);
    acc3 = __builtin_amdgcn_wmma_f32_16x16x32_bf16(false, a, false, b3, (short)0, acc3, false, false);
  }
}

// ---------------------------------------------------------------------------
// Generic bf16 WMMA GEMM:  C[M][N] = sum_sections A_s @ W_s^T
// One wave computes a 16(M) x 64(N) tile: A-fragment reused across 4 WMMAs.
// Block = 128 threads = 4 waves spanning 256 N-columns. grid=(N/256, M/16).
// act: 0 = none, 1 = tanh.  Cf (f32, optionally NT-stored) / Cb (bf16) optional.
// ---------------------------------------------------------------------------
__global__ void __launch_bounds__(128)
gemm_bf16_wmma(GemmSect s0, GemmSect s1, GemmSect s2, int nsect,
               float* __restrict__ Cf, int ldc,
               bf16_t* __restrict__ Cb, int ldcb, int act, int ntstore)
{
  const int lane  = threadIdx.x & 31;
  const int wave  = threadIdx.x >> 5;
  const int half  = lane >> 4;       // 0: lanes 0-15, 1: lanes 16-31
  const int l16   = lane & 15;
  const int mtile = blockIdx.y << 4;
  const int ncol0 = (blockIdx.x * 4 + wave) * 64;
  const int row   = mtile + l16;

  v8f acc0 = {}; v8f acc1 = {}; v8f acc2 = {}; v8f acc3 = {};

  gemm_section(s0, row, l16, half, ncol0, acc0, acc1, acc2, acc3);
  if (nsect > 1) gemm_section(s1, row, l16, half, ncol0, acc0, acc1, acc2, acc3);
  if (nsect > 2) gemm_section(s2, row, l16, half, ncol0, acc0, acc1, acc2, acc3);

  // D layout: VGPR v -> row mtile + v + 8*half, col = l16 within 16-wide tile
#pragma unroll
  for (int nb = 0; nb < 4; ++nb) {
    v8f acc = (nb == 0) ? acc0 : (nb == 1) ? acc1 : (nb == 2) ? acc2 : acc3;
    const int coln = ncol0 + nb * 16 + l16;
#pragma unroll
    for (int v = 0; v < 8; ++v) {
      const int rowd = mtile + v + 8 * half;
      float val = acc[v];
      if (act) val = tanhf(val);
      if (Cf) {
        g_f32 dst = (g_f32)(Cf + (size_t)rowd * ldc + coln);
        if (ntstore) __builtin_nontemporal_store(val, dst);
        else         *dst = val;
      }
      if (Cb) *(g_bf16)(Cb + (size_t)rowd * ldcb + coln) = (bf16_t)val;
    }
  }
}

// ---------------------------------------------------------------------------
// LSTM elementwise cell: gates (64x2048 f32, [i|f|g|o]) + biases -> c, h
// ---------------------------------------------------------------------------
__global__ void __launch_bounds__(256)
lstm_cell_kernel(const float* __restrict__ gates,
                 const float* __restrict__ b_ih,
                 const float* __restrict__ b_hh,
                 float* __restrict__ c,
                 float* __restrict__ h_f,
                 bf16_t* __restrict__ h_b)
{
  const int idx = blockIdx.x * 256 + threadIdx.x;  // 0..BATCH*HID-1
  const int b = idx >> 9, j = idx & 511;
  gc_f32 g   = (gc_f32)(gates + (size_t)b * 2048);
  gc_f32 bi  = (gc_f32)b_ih;
  gc_f32 bh  = (gc_f32)b_hh;
  float ig = g[j]        + bi[j]        + bh[j];
  float fg = g[512 + j]  + bi[512 + j]  + bh[512 + j];
  float gg = g[1024 + j] + bi[1024 + j] + bh[1024 + j];
  float og = g[1536 + j] + bi[1536 + j] + bh[1536 + j];
  ig = 1.f / (1.f + expf(-ig));
  fg = 1.f / (1.f + expf(-fg));
  og = 1.f / (1.f + expf(-og));
  gg = tanhf(gg);
  const float cn = fg * ((gc_f32)c)[idx] + ig * gg;
  const float hn = og * tanhf(cn);
  *(g_f32)(c + idx)   = cn;
  *(g_f32)(h_f + idx) = hn;
  *(g_bf16)(h_b + idx) = (bf16_t)hn;
}

// ---------------------------------------------------------------------------
// Attention: one block per batch row. scores = enc_proj[b] @ h[b], masked
// softmax over S, then a_t = alpha @ enc_hidden[b]. Operands are L2-resident.
// ---------------------------------------------------------------------------
__global__ void __launch_bounds__(256)
attention_kernel(const float* __restrict__ enc_proj,     // [B][S][HID]
                 const float* __restrict__ enc_hid,      // [B][S][2*HID]
                 const unsigned char* __restrict__ mask, // [B][S]
                 const float* __restrict__ h,            // [B][HID]
                 bf16_t* __restrict__ a_bf)              // [B][2*HID]
{
  __shared__ float sh_h[HID];
  __shared__ float sc[SEQ];
  __shared__ float part[256];
  __shared__ float red[64];
  const int b = blockIdx.x;
  const int tid = threadIdx.x;
  for (int i = tid; i < HID; i += 256) sh_h[i] = ((gc_f32)h)[(size_t)b * HID + i];
  __syncthreads();
  { // two threads per score, each covers half of the 512-dot
    const int s = tid >> 1, p = tid & 1;
    gc_f32 ep = (gc_f32)(enc_proj + ((size_t)b * SEQ + s) * HID + p * 256);
    const float* hh = sh_h + p * 256;
    float acc = 0.f;
#pragma unroll 4
    for (int k = 0; k < 256; ++k) acc += ep[k] * hh[k];
    part[tid] = acc;
  }
  __syncthreads();
  if (tid < SEQ) {
    float v = part[2 * tid] + part[2 * tid + 1];
    sc[tid] = mask[(size_t)b * SEQ + tid] ? -3.0e38f : v;
  }
  __syncthreads();
  if (tid < 64) red[tid] = fmaxf(sc[tid], sc[tid + 64]);
  __syncthreads();
  for (int o = 32; o >= 1; o >>= 1) {
    if (tid < o) red[tid] = fmaxf(red[tid], red[tid + o]);
    __syncthreads();
  }
  const float m = red[0];
  __syncthreads();
  if (tid < SEQ) sc[tid] = expf(sc[tid] - m);
  __syncthreads();
  if (tid < 64) red[tid] = sc[tid] + sc[tid + 64];
  __syncthreads();
  for (int o = 32; o >= 1; o >>= 1) {
    if (tid < o) red[tid] += red[tid + o];
    __syncthreads();
  }
  const float inv = 1.0f / red[0];
  __syncthreads();
  if (tid < SEQ) sc[tid] *= inv;
  __syncthreads();
  // a_t: 4 contiguous k per thread (coalesced 16B loads across the wave)
  const int k0 = tid * 4;
  float a0 = 0.f, a1 = 0.f, a2 = 0.f, a3 = 0.f;
  for (int s = 0; s < SEQ; ++s) {
    const float al = sc[s];
    const fvec4 eh = *(gc_f4)(enc_hid + ((size_t)b * SEQ + s) * 1024 + k0);
    a0 += al * eh.x; a1 += al * eh.y; a2 += al * eh.z; a3 += al * eh.w;
  }
  g_bf16 ap = (g_bf16)(a_bf + (size_t)b * 1024 + k0);
  ap[0] = (bf16_t)a0; ap[1] = (bf16_t)a1; ap[2] = (bf16_t)a2; ap[3] = (bf16_t)a3;
}

// ---------------------------------------------------------------------------
// Small utility kernels
// ---------------------------------------------------------------------------
__global__ void cvt_f32_to_bf16(const float* __restrict__ x, bf16_t* __restrict__ y, int n)
{
  const int i = blockIdx.x * 256 + threadIdx.x;
  if (i < n) *(g_bf16)(y + i) = (bf16_t)((gc_f32)x)[i];
}

__global__ void zero_bf16_kernel(bf16_t* __restrict__ y, int n)
{
  const int i = blockIdx.x * 256 + threadIdx.x;
  if (i < n) *(g_bf16)(y + i) = (bf16_t)0.0f;
}

// Y_bf[row][:] = bf16(embed[target[row]][:]), row = t*64+b, 2048 rows
__global__ void __launch_bounds__(256)
embed_gather(const int* __restrict__ target, const float* __restrict__ embed,
             bf16_t* __restrict__ Y)
{
  const int row = blockIdx.x;
  const int tok = target[row];
  gc_f32 e = (gc_f32)(embed + (size_t)tok * HID);
  bf16_t* y = Y + (size_t)row * HID;
  for (int i = threadIdx.x; i < HID; i += 256) *(g_bf16)(y + i) = (bf16_t)e[i];
}

// ---------------------------------------------------------------------------
extern "C" void kernel_launch(void* const* d_in, const int* in_sizes, int n_in,
                              void* d_out, int out_size, void* d_ws, size_t ws_size,
                              hipStream_t stream)
{
  (void)in_sizes; (void)n_in; (void)out_size; (void)ws_size;
  const int*   target   = (const int*)d_in[0];
  const float* enc_hid  = (const float*)d_in[1];
  const float* h0       = (const float*)d_in[2];
  const float* c0       = (const float*)d_in[3];
  const unsigned char* enc_mask = (const unsigned char*)d_in[4];
  const float* embed    = (const float*)d_in[5];
  const float* W_ih     = (const float*)d_in[6];
  const float* W_hh     = (const float*)d_in[7];
  const float* b_ih     = (const float*)d_in[8];
  const float* b_hh     = (const float*)d_in[9];
  const float* W_att    = (const float*)d_in[10];
  const float* W_v      = (const float*)d_in[11];
  const float* W_out    = (const float*)d_in[12];
  float* P = (float*)d_out;

  // ---- workspace carve-out (fixed offsets -> deterministic) ----
  char* ws = (char*)d_ws;
  size_t off = 0;
  auto carve = [&](size_t bytes) -> char* {
    char* p = ws + off; off += (bytes + 255) & ~(size_t)255; return p;
  };
  bf16_t* ehid_bf = (bf16_t*)carve((size_t)BATCH * SEQ * 1024 * 2);   // 16.8 MB
  bf16_t* Watt_bf = (bf16_t*)carve((size_t)HID * 1024 * 2);
  bf16_t* Wih_bf  = (bf16_t*)carve((size_t)2048 * 1024 * 2);
  bf16_t* Whh_bf  = (bf16_t*)carve((size_t)2048 * HID * 2);
  bf16_t* Wv_bf   = (bf16_t*)carve((size_t)HID * 1536 * 2);
  bf16_t* Wout_bf = (bf16_t*)carve((size_t)VOCABN * HID * 2);         // 32.8 MB
  bf16_t* Y_bf    = (bf16_t*)carve((size_t)TSTEP * BATCH * HID * 2);
  bf16_t* O_bf    = (bf16_t*)carve((size_t)TSTEP * BATCH * HID * 2);
  bf16_t* zero_bf = (bf16_t*)carve((size_t)BATCH * HID * 2);
  bf16_t* h_bf    = (bf16_t*)carve((size_t)BATCH * HID * 2);
  bf16_t* a_bf    = (bf16_t*)carve((size_t)BATCH * 1024 * 2);
  float*  encproj = (float*)carve((size_t)BATCH * SEQ * HID * 4);     // 16.8 MB
  float*  gates   = (float*)carve((size_t)BATCH * 2048 * 4);
  float*  c_st    = (float*)carve((size_t)BATCH * HID * 4);
  float*  h_st    = (float*)carve((size_t)BATCH * HID * 4);

  const GemmSect sZ = {nullptr, 0, 0, nullptr, 0, 0, 0};
  auto cvt = [&](const float* x, bf16_t* y, int n) {
    cvt_f32_to_bf16<<<(n + 255) / 256, 256, 0, stream>>>(x, y, n);
  };

  // ---- one-time setup ----
  cvt(enc_hid, ehid_bf, BATCH * SEQ * 1024);
  cvt(W_att, Watt_bf, HID * 1024);
  cvt(W_ih,  Wih_bf,  2048 * 1024);
  cvt(W_hh,  Whh_bf,  2048 * HID);
  cvt(W_v,   Wv_bf,   HID * 1536);
  cvt(W_out, Wout_bf, VOCABN * HID);
  cvt(h0,    h_bf,    BATCH * HID);
  zero_bf16_kernel<<<(BATCH * HID + 255) / 256, 256, 0, stream>>>(zero_bf, BATCH * HID);
  (void)hipMemcpyAsync(c_st, c0, (size_t)BATCH * HID * 4, hipMemcpyDeviceToDevice, stream);
  embed_gather<<<TSTEP * BATCH, 256, 0, stream>>>(target, embed, Y_bf);

  // enc_proj[8192][512] = enc_hidden[8192][1024] @ W_att^T
  {
    GemmSect s0 = {ehid_bf, 1024, 0, Watt_bf, 1024, 0, 1024};
    gemm_bf16_wmma<<<dim3(HID / 256, (BATCH * SEQ) / 16), 128, 0, stream>>>(
        s0, sZ, sZ, 1, encproj, HID, nullptr, 0, 0, 0);
  }

  // ---- sequential decoder scan ----
  for (int t = 0; t < TSTEP; ++t) {
    const bf16_t* oprev = (t == 0) ? zero_bf : (O_bf + (size_t)(t - 1) * BATCH * HID);
    // gates[64][2048] = [y_t|o_prev] @ W_ih^T + h @ W_hh^T  (3 K-sections)
    GemmSect g0 = {Y_bf + (size_t)t * BATCH * HID, HID, 0, Wih_bf, 1024, 0,   HID};
    GemmSect g1 = {oprev,                          HID, 0, Wih_bf, 1024, HID, HID};
    GemmSect g2 = {h_bf,                           HID, 0, Whh_bf, HID,  0,   HID};
    gemm_bf16_wmma<<<dim3(2048 / 256, BATCH / 16), 128, 0, stream>>>(
        g0, g1, g2, 3, gates, 2048, nullptr, 0, 0, 0);

    lstm_cell_kernel<<<(BATCH * HID) / 256, 256, 0, stream>>>(
        gates, b_ih, b_hh, c_st, h_st, h_bf);

    attention_kernel<<<BATCH, 256, 0, stream>>>(encproj, enc_hid, enc_mask, h_st, a_bf);

    // o_t[64][512] = tanh([a_t|h] @ W_v^T), written bf16 into O (also next o_prev)
    GemmSect v0 = {a_bf, 1024, 0, Wv_bf, 1536, 0,    1024};
    GemmSect v1 = {h_bf, HID,  0, Wv_bf, 1536, 1024, HID};
    gemm_bf16_wmma<<<dim3(HID / 256, BATCH / 16), 128, 0, stream>>>(
        v0, v1, sZ, 2, nullptr, 0, O_bf + (size_t)t * BATCH * HID, HID, 1, 0);
  }

  // P[2048][32000] = O @ W_out^T  (dominant GEMM; NT stores keep W_out in L2)
  {
    GemmSect f0 = {O_bf, HID, 0, Wout_bf, HID, 0, HID};
    gemm_bf16_wmma<<<dim3(VOCABN / 256, (TSTEP * BATCH) / 16), 128, 0, stream>>>(
        f0, sZ, sZ, 1, P, VOCABN, nullptr, 0, 0, 1);
  }
}
